// _SelfAttentionBlock_2585570312672
// MI455X (gfx1250) — matrix operand order repaired and checked by reading, not compiled, not run on hardware
//
#include <hip/hip_runtime.h>
#include <hip/hip_bf16.h>
#include <math.h>

// ---------------------------------------------------------------------------
// CDNA5 (gfx1250) fused SelfAttentionBlock.
//   y[512,4096]   = (BNfold(Wk)|BNfold(Wv)) x x(f32->f16)  (WMMA GEMM + bias)
//   kf,vf         = FReLU(y) (depthwise3x3 + BN + max)     (stencil, f16 out)
//   sim[256,256]  = (vf x kf^T) * N^-0.5                   (WMMA GEMM, transB)
//   simS          = softmax_k(sim)                         (wave32 row softmax)
//   M2[512,256]   = w_w x simS^T  (f16 out)                (WMMA GEMM, transB)
//   out[512,4096] = M2 x vf + w_b                          (WMMA GEMM + bias)
// ---------------------------------------------------------------------------

typedef _Float16 h8   __attribute__((ext_vector_type(8)));
typedef _Float16 v16h __attribute__((ext_vector_type(16)));
typedef float    v8f  __attribute__((ext_vector_type(8)));
typedef float    f4v  __attribute__((ext_vector_type(4)));

#define BM 128
#define BN 128
#define BK 32
#define LDSS 40   // f16 elems per LDS row (32 + 8 pad), 80B keeps 16B alignment

// A fragment (16x32 f16, MxK): lane<16 holds K {0..7},{16..23}; lane>=16 {8..15},{24..31}
__device__ __forceinline__ v16h load_frag_a(const _Float16* row, int lhalf) {
    h8 lo = *reinterpret_cast<const h8*>(row + lhalf * 8);
    h8 hi = *reinterpret_cast<const h8*>(row + 16 + lhalf * 8);
    return __builtin_shufflevector(lo, hi, 0,1,2,3,4,5,6,7,8,9,10,11,12,13,14,15);
}
// B fragment (32x16 f16, KxN): lane = column n%16; lanes<16 hold K {0..7},{16..23}; lanes>=16 {8..15},{24..31}
__device__ __forceinline__ v16h load_frag_b(const _Float16* row, int lhalf) {
    h8 lo = *reinterpret_cast<const h8*>(row + lhalf * 8);
    h8 hi = *reinterpret_cast<const h8*>(row + 16 + lhalf * 8);
    return __builtin_shufflevector(lo, hi, 0,1,2,3,4,5,6,7,8,9,10,11,12,13,14,15);
}

// C = alpha * A[M,K] x B (+ bias[m]).  TRANS_B: B given as BT[N,K] row-major f16.
// !TRANS_B: B is [K,N] row-major, f16 or f32 (B_F32, converted during LDS store).
// C_F16: emit _Float16 output, else float.  M%128==0, N%128==0, K%32==0 assumed.
template <bool TRANS_B, bool B_F32, bool HAS_BIAS, bool C_F16>
__global__ __launch_bounds__(256) void gemm_wmma(
    const _Float16* __restrict__ A, const void* __restrict__ Bv,
    void* __restrict__ Cv, const float* __restrict__ bias,
    int M, int N, int K, float alpha,
    long strideA, long strideB, long strideC)
{
    static_assert(!(TRANS_B && B_F32), "f32 B only supported in non-transposed path");
    __shared__ __align__(16) _Float16 As[2][BM * LDSS];
    __shared__ __align__(16) _Float16 Bs[2][BN * LDSS];

    const int tid  = threadIdx.x;
    const int wid  = tid >> 5;
    const int lane = tid & 31;
    const int lhalf = lane >> 4;
    const int l16   = lane & 15;

    const int m0 = blockIdx.y * BM;
    const int n0 = blockIdx.x * BN;
    A += (long)blockIdx.z * strideA;
    const _Float16* Bh = B_F32 ? nullptr : (const _Float16*)Bv + (long)blockIdx.z * strideB;
    const float*    Bf = B_F32 ? (const float*)Bv + (long)blockIdx.z * strideB : nullptr;
    float*    Cf = C_F16 ? nullptr : (float*)Cv + (long)blockIdx.z * strideC;
    _Float16* Ch = C_F16 ? (_Float16*)Cv + (long)blockIdx.z * strideC : nullptr;

    const int waveM = (wid >> 1) * 32;  // 0,32,64,96
    const int waveN = (wid & 1) * 64;   // 0,64

    v8f acc[2][4] = {};

    // copy-in indexing: A tile row r = tid/2, 16-elem half h = (tid&1)*16
    const int ar = tid >> 1;
    const int ah = (tid & 1) * 16;
    // B (non-trans) tile: k-row = tid&31, 16-wide n chunk = (tid>>5)*16
    const int bk = tid & 31;
    const int bn = (tid >> 5) * 16;

    uint4 sa0, sa1, sb0, sb1, sb2, sb3;   // staged global data

    auto gload = [&](int k0) {
        const uint4* gA = reinterpret_cast<const uint4*>(A + (long)(m0 + ar) * K + k0 + ah);
        sa0 = gA[0]; sa1 = gA[1];
        if constexpr (TRANS_B) {
            const uint4* gB = reinterpret_cast<const uint4*>(Bh + (long)(n0 + ar) * K + k0 + ah);
            sb0 = gB[0]; sb1 = gB[1];
        } else if constexpr (B_F32) {
            const uint4* gB = reinterpret_cast<const uint4*>(Bf + (long)(k0 + bk) * N + n0 + bn);
            sb0 = gB[0]; sb1 = gB[1]; sb2 = gB[2]; sb3 = gB[3];
        } else {
            const uint4* gB = reinterpret_cast<const uint4*>(Bh + (long)(k0 + bk) * N + n0 + bn);
            sb0 = gB[0]; sb1 = gB[1];
        }
    };

    auto lstore = [&](int buf) {
        _Float16* as = &As[buf][0];
        _Float16* bs = &Bs[buf][0];
        *reinterpret_cast<uint4*>(as + ar * LDSS + ah)     = sa0;
        *reinterpret_cast<uint4*>(as + ar * LDSS + ah + 8) = sa1;
        if constexpr (TRANS_B) {
            *reinterpret_cast<uint4*>(bs + ar * LDSS + ah)     = sb0;
            *reinterpret_cast<uint4*>(bs + ar * LDSS + ah + 8) = sb1;
        } else if constexpr (B_F32) {
            f4v f0 = __builtin_bit_cast(f4v, sb0);
            f4v f1 = __builtin_bit_cast(f4v, sb1);
            f4v f2 = __builtin_bit_cast(f4v, sb2);
            f4v f3 = __builtin_bit_cast(f4v, sb3);
#pragma unroll
            for (int j = 0; j < 4; ++j) {
                bs[(bn + j)      * LDSS + bk] = (_Float16)f0[j];
                bs[(bn + 4 + j)  * LDSS + bk] = (_Float16)f1[j];
                bs[(bn + 8 + j)  * LDSS + bk] = (_Float16)f2[j];
                bs[(bn + 12 + j) * LDSS + bk] = (_Float16)f3[j];
            }
        } else {
            h8 t0 = __builtin_bit_cast(h8, sb0);
            h8 t1 = __builtin_bit_cast(h8, sb1);
#pragma unroll
            for (int j = 0; j < 8; ++j) {
                bs[(bn + j)     * LDSS + bk] = t0[j];
                bs[(bn + 8 + j) * LDSS + bk] = t1[j];
            }
        }
    };

    const int nk = K / BK;
    gload(0);
    lstore(0);
    __syncthreads();

    for (int t = 0; t < nk; ++t) {
        const int cur = t & 1;
        if (t + 1 < nk) gload((t + 1) * BK);

        const _Float16* as = &As[cur][0];
        const _Float16* bsr = &Bs[cur][0];
        v16h afrag[2], bfrag[4];
#pragma unroll
        for (int i = 0; i < 2; ++i)
            afrag[i] = load_frag_a(as + (waveM + i * 16 + l16) * LDSS, lhalf);
#pragma unroll
        for (int j = 0; j < 4; ++j)
            bfrag[j] = load_frag_b(bsr + (waveN + j * 16 + l16) * LDSS, lhalf);
#pragma unroll
        for (int i = 0; i < 2; ++i)
#pragma unroll
            for (int j = 0; j < 4; ++j)
                acc[i][j] = __builtin_amdgcn_wmma_f32_16x16x32_f16(
                    false, afrag[i], false, bfrag[j], (short)0, acc[i][j], false, false);

        if (t + 1 < nk) {
            lstore(cur ^ 1);      // other buffer: no race with readers of `cur`
            __syncthreads();      // single barrier per k-step
        }
    }

    // hoisted bias (no branches in the store loop)
    float bvals[2][8];
    if constexpr (HAS_BIAS) {
#pragma unroll
        for (int i = 0; i < 2; ++i)
#pragma unroll
            for (int r = 0; r < 8; ++r)
                bvals[i][r] = bias[m0 + waveM + i * 16 + lhalf * 8 + r];
    }

    // D layout (16x16 f32): lanes<16 -> M=r, N=lane; lanes>=16 -> M=r+8, N=lane-16
#pragma unroll
    for (int i = 0; i < 2; ++i) {
#pragma unroll
        for (int j = 0; j < 4; ++j) {
            const int mb = m0 + waveM + i * 16 + lhalf * 8;
            const int nb = n0 + waveN + j * 16 + l16;
#pragma unroll
            for (int r = 0; r < 8; ++r) {
                float v = acc[i][j][r] * alpha;
                if constexpr (HAS_BIAS) v += bvals[i][r];
                if constexpr (C_F16) Ch[(long)(mb + r) * N + nb] = (_Float16)v;
                else                 Cf[(long)(mb + r) * N + nb] = v;
            }
        }
    }
}

// ---------------------------------------------------------------------------
// Pack folded branch weights: rows 0..255 key, 256..511 value.
// Wpk[r][c] = cw[r][c]*s_r (f16);  biaspk[r] = cb*s + (bb - bm*s)
__global__ __launch_bounds__(256) void pack_weights(
    const float* fk_cw, const float* fk_cb, const float* fk_bg, const float* fk_bb,
    const float* fk_bm, const float* fk_bv,
    const float* fv_cw, const float* fv_cb, const float* fv_bg, const float* fv_bb,
    const float* fv_bm, const float* fv_bv,
    _Float16* __restrict__ Wpk, float* __restrict__ biaspk)
{
    const int r = blockIdx.x;
    const bool isK = r < 256;
    const int rr = isK ? r : r - 256;
    const float* cw = (isK ? fk_cw : fv_cw) + (long)rr * 512;
    const float g = (isK ? fk_bg : fv_bg)[rr];
    const float v = (isK ? fk_bv : fv_bv)[rr];
    const float s = g * rsqrtf(v + 1e-5f);
    for (int c = threadIdx.x; c < 512; c += 256)
        Wpk[(long)r * 512 + c] = (_Float16)(cw[c] * s);
    if (threadIdx.x == 0) {
        const float cb = (isK ? fk_cb : fv_cb)[rr];
        const float bb = (isK ? fk_bb : fv_bb)[rr];
        const float bm = (isK ? fk_bm : fv_bm)[rr];
        biaspk[r] = cb * s + (bb - bm * s);
    }
}

__global__ __launch_bounds__(256) void f32_to_f16(
    const float* __restrict__ in, _Float16* __restrict__ out, long n)
{
    long i = (long)blockIdx.x * blockDim.x + threadIdx.x;
    const long stride = (long)gridDim.x * blockDim.x;
    for (; i < n; i += stride) out[i] = (_Float16)in[i];
}

// FReLU: out = max(y, BN(dwconv3x3(y))) for both branches, f16 row-major output.
__global__ __launch_bounds__(256) void frelu_kernel(
    const float* __restrict__ y,                       // [B,512,64,64]
    const float* fk_dw, const float* fk_db, const float* fk_dg, const float* fk_dbb,
    const float* fk_dm, const float* fk_dv,
    const float* fv_dw, const float* fv_db, const float* fv_dg, const float* fv_dbb,
    const float* fv_dm, const float* fv_dv,
    _Float16* __restrict__ kf, _Float16* __restrict__ vf)  // each [B,256,4096]
{
    const int p = blockIdx.x * 256 + threadIdx.x;      // 0..4095
    const int c = blockIdx.y;                          // 0..511
    const int b = blockIdx.z;
    const int h = p >> 6, w = p & 63;
    const float* yp = y + ((long)b * 512 + c) * 4096;

    const bool isK = c < 256;
    const int cc = isK ? c : c - 256;
    const float* dwp = (isK ? fk_dw : fv_dw) + cc * 9;
    const float db = (isK ? fk_db : fv_db)[cc];
    const float g  = (isK ? fk_dg : fv_dg)[cc];
    const float bb = (isK ? fk_dbb : fv_dbb)[cc];
    const float mm = (isK ? fk_dm : fv_dm)[cc];
    const float vv = (isK ? fk_dv : fv_dv)[cc];
    const float s = g * rsqrtf(vv + 1e-5f);
    const float t = bb - mm * s;

    float conv = 0.f;
#pragma unroll
    for (int dh = -1; dh <= 1; ++dh)
#pragma unroll
        for (int dw2 = -1; dw2 <= 1; ++dw2) {
            const int hh = h + dh, ww = w + dw2;
            const float xv = (hh >= 0 && hh < 64 && ww >= 0 && ww < 64) ? yp[hh * 64 + ww] : 0.f;
            conv += xv * dwp[(dh + 1) * 3 + (dw2 + 1)];
        }
    const float bnv = (conv + db) * s + t;
    const float out = fmaxf(yp[p], bnv);
    _Float16* dst = (isK ? kf : vf) + ((long)b * 256 + cc) * 4096 + p;
    *dst = (_Float16)out;
}

// Row softmax over 256 elems; one wave32 per row; f16 output (GEMM operand).
__global__ __launch_bounds__(256) void softmax_kernel(
    const float* __restrict__ simp, _Float16* __restrict__ simh)
{
    const int wid = threadIdx.x >> 5, lane = threadIdx.x & 31;
    const long row = (long)blockIdx.x * 8 + wid;       // B*256 = 2048 rows
    const float* rp = simp + row * 256;
    float vals[8];
    float mx = -3.4e38f;
#pragma unroll
    for (int i = 0; i < 8; ++i) { vals[i] = rp[i * 32 + lane]; mx = fmaxf(mx, vals[i]); }
#pragma unroll
    for (int off = 16; off > 0; off >>= 1) mx = fmaxf(mx, __shfl_xor(mx, off, 32));
    float sum = 0.f;
#pragma unroll
    for (int i = 0; i < 8; ++i) { vals[i] = __expf(vals[i] - mx); sum += vals[i]; }
#pragma unroll
    for (int off = 16; off > 0; off >>= 1) sum += __shfl_xor(sum, off, 32);
    const float inv = 1.f / sum;
#pragma unroll
    for (int i = 0; i < 8; ++i)
        simh[row * 256 + i * 32 + lane] = (_Float16)(vals[i] * inv);
}

// ---------------------------------------------------------------------------
extern "C" void kernel_launch(void* const* d_in, const int* in_sizes, int n_in,
                              void* d_out, int out_size, void* d_ws, size_t ws_size,
                              hipStream_t stream)
{
    (void)in_sizes; (void)n_in; (void)out_size; (void)ws_size;
    const float* x      = (const float*)d_in[0];
    const float* fk_cw  = (const float*)d_in[1];
    const float* fk_cb  = (const float*)d_in[2];
    const float* fk_bg  = (const float*)d_in[3];
    const float* fk_bb  = (const float*)d_in[4];
    const float* fk_bm  = (const float*)d_in[5];
    const float* fk_bv  = (const float*)d_in[6];
    const float* fk_dw  = (const float*)d_in[7];
    const float* fk_db  = (const float*)d_in[8];
    const float* fk_dg  = (const float*)d_in[9];
    const float* fk_dbb = (const float*)d_in[10];
    const float* fk_dm  = (const float*)d_in[11];
    const float* fk_dv  = (const float*)d_in[12];
    const float* fv_cw  = (const float*)d_in[13];
    const float* fv_cb  = (const float*)d_in[14];
    const float* fv_bg  = (const float*)d_in[15];
    const float* fv_bb  = (const float*)d_in[16];
    const float* fv_bm  = (const float*)d_in[17];
    const float* fv_bv  = (const float*)d_in[18];
    const float* fv_dw  = (const float*)d_in[19];
    const float* fv_db  = (const float*)d_in[20];
    const float* fv_dg  = (const float*)d_in[21];
    const float* fv_dbb = (const float*)d_in[22];
    const float* fv_dm  = (const float*)d_in[23];
    const float* fv_dv  = (const float*)d_in[24];
    const float* w_w    = (const float*)d_in[25];
    const float* w_b    = (const float*)d_in[26];
    float* out = (float*)d_out;

    constexpr int  Bb = 8;
    constexpr long N  = 4096;      // H*W
    constexpr long C  = 512;
    constexpr long Kc = 256, Vc = 256, O = 512;

    // workspace carve-up
    char* ws = (char*)d_ws;
    size_t off = 0;
    auto carve = [&](size_t bytes) -> char* {
        char* p = ws + off;
        off = (off + bytes + 255) & ~(size_t)255;
        return p;
    };
    _Float16* Wpk    = (_Float16*)carve(512 * 512 * 2);
    float*    biaspk = (float*)   carve(512 * 4);
    float*    y      = (float*)   carve((size_t)Bb * 512 * N * 4);   // 64 MB
    _Float16* kf     = (_Float16*)carve((size_t)Bb * Kc * N * 2);    // 16 MB
    _Float16* vf     = (_Float16*)carve((size_t)Bb * Vc * N * 2);    // 16 MB
    float*    simp   = (float*)   carve((size_t)Bb * Vc * Kc * 4);
    _Float16* simh   = (_Float16*)carve((size_t)Bb * Vc * Kc * 2);
    _Float16* ww_h   = (_Float16*)carve(O * Kc * 2);
    _Float16* M2h    = (_Float16*)carve((size_t)Bb * O * Vc * 2);

    // 1) small weight preps
    f32_to_f16<<<512, 256, 0, stream>>>(w_w, ww_h, O * Kc);
    pack_weights<<<512, 256, 0, stream>>>(fk_cw, fk_cb, fk_bg, fk_bb, fk_bm, fk_bv,
                                          fv_cw, fv_cb, fv_bg, fv_bb, fv_bm, fv_bv,
                                          Wpk, biaspk);

    // 2) branch GEMM: y[512,4096] = Wpk[512,512] x x_f32[512,4096] + biaspk
    //    (x loaded f32 straight from HBM/L2, converted to f16 during LDS store)
    gemm_wmma<false, true, true, false><<<dim3(N / BN, 512 / BM, Bb), 256, 0, stream>>>(
        Wpk, x, y, biaspk, 512, (int)N, 512, 1.0f,
        0L, C * N, 512L * N);

    // 3) FReLU -> kf/vf (f16)
    frelu_kernel<<<dim3(16, 512, Bb), 256, 0, stream>>>(
        y, fk_dw, fk_db, fk_dg, fk_dbb, fk_dm, fk_dv,
        fv_dw, fv_db, fv_dg, fv_dbb, fv_dm, fv_dv, kf, vf);

    // 4) sim = (vf x kf^T) / 64
    gemm_wmma<true, false, false, false><<<dim3(Kc / BN, Vc / BM, Bb), 256, 0, stream>>>(
        vf, kf, simp, nullptr, (int)Vc, (int)Kc, (int)N, 1.0f / 64.0f,
        Vc * N, Kc * N, Vc * Kc);

    // 5) softmax over k
    softmax_kernel<<<(Bb * 256) / 8, 256, 0, stream>>>(simp, simh);

    // 6) M2 = w_w x sim^T  (reassociated context+projection), f16 output direct
    gemm_wmma<true, false, false, true><<<dim3(Vc / BN, O / BM, Bb), 256, 0, stream>>>(
        ww_h, simh, M2h, nullptr, (int)O, (int)Vc, (int)Kc, 1.0f,
        0L, Vc * Kc, O * Vc);

    // 7) out = M2 x vf + w_b
    gemm_wmma<false, false, true, false><<<dim3(N / BN, O / BM, Bb), 256, 0, stream>>>(
        M2h, vf, out, w_b, (int)O, (int)N, (int)Vc, 1.0f,
        O * Vc, Vc * N, O * N);
}
